// DiffusionMACE_21809843929211
// MI455X (gfx1250) — matrix operand order repaired and checked by reading, not compiled, not run on hardware
//
#include <hip/hip_runtime.h>
#include <math.h>

#define DEV __device__ __forceinline__

typedef __attribute__((ext_vector_type(16))) _Float16 v16h;
typedef __attribute__((ext_vector_type(8)))  _Float16 v8h;
typedef __attribute__((ext_vector_type(8)))  float    v8f;

union AFrag { v16h v; v8h p[2]; _Float16 h[16]; };
union CFrag { v8f  v; float    f[8];  };

// l(k) for the 16 lm channels: [0,1,1,1,2,2,2,2,2,3,3,3,3,3,3,3]
DEV int lof(int k) { return (k >= 9) ? 3 : (k >= 4) ? 2 : (k >= 1) ? 1 : 0; }

DEV float siluf(float x) { return x / (1.0f + __expf(-x)); }

// ---------------------------------------------------------------------------
// Fragment layout (CDNA5 ISA 7.12.2), wave32, 16x16x32 f16:
//  A: lane L, row m=L&15, elements e<8 -> k = 8*(L>=16)+e ; e>=8 -> k = 16+8*(L>=16)+(e-8)
//     => two contiguous v8h (b128) loads from a f16 row-major X.
//  B: lane L, col n=L&15, element e -> same k as A (table order).  Weights are repacked
//     into fragment order so a B fragment is two contiguous v8h loads.
// Packed weight layout: Wp[ ((tile*KS + ks)*32 + lane)*16 + e ]
// ---------------------------------------------------------------------------
__global__ void repack_w(const float* __restrict__ W, _Float16* __restrict__ Wp,
                         int K, int NC, int nmat)
{
  const size_t per = (size_t)(NC / 16) * (K / 32) * 512;  // halves per matrix
  const size_t total = per * nmat;
  const size_t tid = (size_t)blockIdx.x * blockDim.x + threadIdx.x;
  if (tid >= total) return;
  const int mat = (int)(tid / per);
  const size_t r = tid - (size_t)mat * per;
  const int e    = (int)(r & 15);
  const int lane = (int)((r >> 4) & 31);
  const int KS   = K >> 5;
  const int ks   = (int)((r >> 9) % KS);
  const int t    = (int)((r >> 9) / KS);
  const int kk   = ks * 32 + (lane >> 4) * 8 + (e & 7) + (e >> 3) * 16;   // b_kidx
  const int n    = t * 16 + (lane & 15);
  Wp[tid] = (_Float16)W[(size_t)mat * K * NC + (size_t)kk * NC + n];
}

// ---------------------------------------------------------------------------
// Generic GEMM, f16 activations: Y = act(X[R,K] @ W[K,NC]).
// One wave = 16x64 tile. grid = (R/16, NC/64), block = 32.
// ---------------------------------------------------------------------------
__global__ __launch_bounds__(32)
void wmma_gemm_h(const _Float16* __restrict__ X, const _Float16* __restrict__ Wp,
                 _Float16* __restrict__ Yh, float* __restrict__ Yf,
                 int K, int NC, int applySilu)
{
  const int lane = threadIdx.x & 31;
  const int m    = lane & 15;
  const int half = lane >> 4;
  const size_t row = (size_t)blockIdx.x * 16 + m;
  const int KS  = K >> 5;
  const int t0  = blockIdx.y * 4;

  CFrag acc[4];
  #pragma unroll
  for (int t = 0; t < 4; ++t)
    #pragma unroll
    for (int g = 0; g < 8; ++g) acc[t].f[g] = 0.0f;

  const _Float16* Xrow = X + row * K + half * 8;
  for (int ks = 0; ks < KS; ++ks) {
    AFrag a;
    a.p[0] = *(const v8h*)(Xrow + ks * 32);
    a.p[1] = *(const v8h*)(Xrow + ks * 32 + 16);
    #pragma unroll
    for (int t = 0; t < 4; ++t) {
      AFrag b;
      const _Float16* bp = Wp + (((size_t)(t0 + t) * KS + ks) * 32 + lane) * 16;
      b.p[0] = *(const v8h*)(bp);
      b.p[1] = *(const v8h*)(bp + 8);
      acc[t].v = __builtin_amdgcn_wmma_f32_16x16x32_f16(
          false, a.v, false, b.v, (short)0, acc[t].v, false, false);
    }
  }

  const size_t orow0 = (size_t)blockIdx.x * 16;
  #pragma unroll
  for (int t = 0; t < 4; ++t) {
    const int col = (t0 + t) * 16 + m;
    #pragma unroll
    for (int g = 0; g < 8; ++g) {
      const size_t r = orow0 + g + half * 8;
      float val = acc[t].f[g];
      if (applySilu) val = siluf(val);
      if (Yh) Yh[r * NC + col] = (_Float16)val;
      if (Yf) Yf[r * NC + col] = val;
    }
  }
}

// ---------------------------------------------------------------------------
// Per-k node GEMM: O[n,k,:] = T[n,k,:] @ W[l(k)] (+ addend). K=64, NC=64.
// T is f16 [N,16,64]; Wp packed per-l (4096 halves each). grid=(N/16,16).
// ---------------------------------------------------------------------------
__global__ __launch_bounds__(32)
void wmma_nodek_h(const _Float16* __restrict__ T, const _Float16* __restrict__ Wp,
                  const float* __restrict__ addend,
                  float* __restrict__ Of, _Float16* __restrict__ Oh)
{
  const int lane = threadIdx.x & 31;
  const int m    = lane & 15;
  const int half = lane >> 4;
  const int k    = blockIdx.y;
  const int l    = lof(k);
  const size_t node0 = (size_t)blockIdx.x * 16;
  const _Float16* Wl = Wp + (size_t)l * 4096;

  CFrag acc[4];
  #pragma unroll
  for (int t = 0; t < 4; ++t)
    #pragma unroll
    for (int g = 0; g < 8; ++g) acc[t].f[g] = 0.0f;

  const _Float16* Trow = T + (node0 + m) * 1024 + k * 64 + half * 8;
  #pragma unroll
  for (int ks = 0; ks < 2; ++ks) {
    AFrag a;
    a.p[0] = *(const v8h*)(Trow + ks * 32);
    a.p[1] = *(const v8h*)(Trow + ks * 32 + 16);
    #pragma unroll
    for (int t = 0; t < 4; ++t) {
      AFrag b;
      const _Float16* bp = Wl + (((size_t)t * 2 + ks) * 32 + lane) * 16;
      b.p[0] = *(const v8h*)(bp);
      b.p[1] = *(const v8h*)(bp + 8);
      acc[t].v = __builtin_amdgcn_wmma_f32_16x16x32_f16(
          false, a.v, false, b.v, (short)0, acc[t].v, false, false);
    }
  }

  #pragma unroll
  for (int t = 0; t < 4; ++t) {
    const int d = t * 16 + m;
    #pragma unroll
    for (int g = 0; g < 8; ++g) {
      const size_t node = node0 + g + half * 8;
      const size_t idx = node * 1024 + k * 64 + d;
      float val = acc[t].f[g];
      if (addend) val += addend[idx];
      if (Of) Of[idx] = val;
      if (Oh) Oh[idx] = (_Float16)val;
    }
  }
}

// ---------------------------------------------------------------------------
// sc GEMM: sc[n,k,d] = A'[k,n,:320] @ scW[l(k)] ; A' = feats x aug precomputed.
// K=320 (kk=c*5+z), NC=64; Wp packed per-l (20480 halves). grid=(N/16,16).
// ---------------------------------------------------------------------------
__global__ __launch_bounds__(32)
void wmma_sc_h(const _Float16* __restrict__ Ap, const _Float16* __restrict__ Wp,
               float* __restrict__ sc, int N)
{
  const int lane = threadIdx.x & 31;
  const int m    = lane & 15;
  const int half = lane >> 4;
  const int k    = blockIdx.y;
  const int l    = lof(k);
  const size_t node0 = (size_t)blockIdx.x * 16;
  const _Float16* Wl = Wp + (size_t)l * 20480;
  const _Float16* Arow = Ap + ((size_t)k * N + node0 + m) * 320 + half * 8;

  CFrag acc[4];
  #pragma unroll
  for (int t = 0; t < 4; ++t)
    #pragma unroll
    for (int g = 0; g < 8; ++g) acc[t].f[g] = 0.0f;

  for (int ks = 0; ks < 10; ++ks) {
    AFrag a;
    a.p[0] = *(const v8h*)(Arow + ks * 32);
    a.p[1] = *(const v8h*)(Arow + ks * 32 + 16);
    #pragma unroll
    for (int t = 0; t < 4; ++t) {
      AFrag b;
      const _Float16* bp = Wl + (((size_t)t * 10 + ks) * 32 + lane) * 16;
      b.p[0] = *(const v8h*)(bp);
      b.p[1] = *(const v8h*)(bp + 8);
      acc[t].v = __builtin_amdgcn_wmma_f32_16x16x32_f16(
          false, a.v, false, b.v, (short)0, acc[t].v, false, false);
    }
  }

  #pragma unroll
  for (int t = 0; t < 4; ++t) {
    const int d = t * 16 + m;
    #pragma unroll
    for (int g = 0; g < 8; ++g) {
      const size_t node = node0 + g + half * 8;
      sc[node * 1024 + k * 64 + d] = acc[t].f[g];
    }
  }
}

// A'[k][n][c*5+z] = feats[n,k,c] * aug[n,z]
__global__ void build_aprime(const _Float16* __restrict__ featsh, const float* __restrict__ aug,
                             _Float16* __restrict__ Ap, int N)
{
  const size_t t = (size_t)blockIdx.x * blockDim.x + threadIdx.x;
  const int c = (int)(t & 63);
  const size_t nk = t >> 6;
  const int n = (int)(nk % N);
  const int k = (int)(nk / N);
  if (k >= 16) return;
  const float f = (float)featsh[(size_t)n * 1024 + k * 64 + c];
  _Float16* o = Ap + ((size_t)k * N + n) * 320 + c * 5;
  #pragma unroll
  for (int z = 0; z < 5; ++z) o[z] = (_Float16)(f * aug[(size_t)n * 5 + z]);
}

__global__ void f32_to_f16(const float* __restrict__ in, _Float16* __restrict__ out, size_t n)
{
  size_t i = (size_t)blockIdx.x * blockDim.x + threadIdx.x;
  const size_t stride = (size_t)gridDim.x * blockDim.x;
  for (; i < n; i += stride) out[i] = (_Float16)in[i];
}

__global__ void zero_kernel(float* __restrict__ p, size_t n)
{
  size_t i = (size_t)blockIdx.x * blockDim.x + threadIdx.x;
  const size_t stride = (size_t)gridDim.x * blockDim.x;
  for (; i < n; i += stride) p[i] = 0.0f;
}

// ---------------------------------------------------------------------------
DEV void diff_mlp(float s,
                  const float* W0, const float* b0,
                  const float* W1, const float* b1,
                  const float* W2, const float* b2,
                  float* out)
{
  float h1[16], h2[16];
  #pragma unroll
  for (int i = 0; i < 16; ++i) h1[i] = siluf(s * W0[i] + b0[i]);
  #pragma unroll
  for (int j = 0; j < 16; ++j) {
    float a = b1[j];
    #pragma unroll
    for (int i = 0; i < 16; ++i) a += h1[i] * W1[i * 16 + j];
    h2[j] = siluf(a);
  }
  #pragma unroll
  for (int j = 0; j < 16; ++j) {
    float a = b2[j];
    #pragma unroll
    for (int i = 0; i < 16; ++i) a += h2[i] * W2[i * 16 + j];
    out[j] = a;
  }
}

__global__ void node_prep(const float* __restrict__ nscal, const float* __restrict__ attrs,
                          const float* W0, const float* b0, const float* W1, const float* b1,
                          const float* W2, const float* b2,
                          const float* __restrict__ amw, float* __restrict__ aug, int N)
{
  const int n = blockIdx.x * blockDim.x + threadIdx.x;
  if (n >= N) return;
  float nd[16];
  diff_mlp(nscal[n], W0, b0, W1, b1, W2, b2, nd);
  float at[5];
  #pragma unroll
  for (int z = 0; z < 5; ++z) at[z] = attrs[(size_t)n * 5 + z];
  #pragma unroll
  for (int kk = 0; kk < 5; ++kk) {
    float a = 0.0f;
    for (int i = 0; i < 16; ++i)
      #pragma unroll
      for (int z = 0; z < 5; ++z)
        a += nd[i] * at[z] * amw[(i * 5 + z) * 5 + kk];
    aug[(size_t)n * 5 + kk] = a * 0.11180339887498949f;  // 1/sqrt(80)
  }
}

__global__ void feats_init(const float* __restrict__ aug, const float* __restrict__ emb,
                           float* __restrict__ feats, _Float16* __restrict__ featsh, int N)
{
  const int t = blockIdx.x * blockDim.x + threadIdx.x;
  const int n = t >> 6, c = t & 63;
  if (n >= N) return;
  float a = 0.0f;
  #pragma unroll
  for (int z = 0; z < 5; ++z) a += aug[(size_t)n * 5 + z] * emb[z * 64 + c];
  const size_t base = (size_t)n * 1024;
  feats[base + c] = a;
  featsh[base + c] = (_Float16)a;
  #pragma unroll
  for (int k = 1; k < 16; ++k) {
    feats[base + k * 64 + c] = 0.0f;
    featsh[base + k * 64 + c] = (_Float16)0.0f;
  }
}

// ---------------------------------------------------------------------------
__global__ __launch_bounds__(256)
void edge_prep(const float* __restrict__ pos, const float* __restrict__ shifts,
               const float* __restrict__ escal,
               const int* __restrict__ src, const int* __restrict__ dst,
               const float* W0, const float* b0, const float* W1, const float* b1,
               const float* W2, const float* b2,
               const float* __restrict__ emw, const float* __restrict__ ehw,
               float* __restrict__ sh_out, _Float16* __restrict__ ef_out, int E)
{
  __shared__ float s_bess[4][8];
  __shared__ float s_ediff[4][16];
  __shared__ float s_ef[4][64];

  const int g  = threadIdx.x & 63;
  const int eg = threadIdx.x >> 6;
  const size_t e = (size_t)blockIdx.x * 4 + eg;
  if (e >= (size_t)E) return;

  if (g == 0) {
    const int si = src[e], di = dst[e];
    const float vx = pos[(size_t)di * 3 + 0] - pos[(size_t)si * 3 + 0] + shifts[e * 3 + 0];
    const float vy = pos[(size_t)di * 3 + 1] - pos[(size_t)si * 3 + 1] + shifts[e * 3 + 1];
    const float vz = pos[(size_t)di * 3 + 2] - pos[(size_t)si * 3 + 2] + shifts[e * 3 + 2];
    const float r = sqrtf(vx * vx + vy * vy + vz * vz);
    const float inv = 1.0f / (r + 1e-8f);
    const float x = vx * inv, y = vy * inv, z = vz * inv;
    const float s3 = 1.7320508075688772f, s5 = 2.23606797749979f, s7 = 2.6457513110645906f,
                s15 = 3.872983346207417f, s42 = 6.48074069840786f,
                s70 = 8.366600265340756f, s105 = 10.246950765959598f;
    float sh[16];
    sh[0] = 1.0f; sh[1] = s3 * x; sh[2] = s3 * y; sh[3] = s3 * z;
    sh[4] = s15 * x * y; sh[5] = s15 * y * z; sh[6] = 0.5f * s5 * (3.0f * z * z - 1.0f);
    sh[7] = s15 * x * z; sh[8] = 0.5f * s15 * (x * x - y * y);
    sh[9]  = 0.25f * s70 * y * (3.0f * x * x - y * y);
    sh[10] = s105 * x * y * z;
    sh[11] = 0.25f * s42 * y * (5.0f * z * z - 1.0f);
    sh[12] = 0.5f * s7 * (5.0f * z * z * z - 3.0f * z);
    sh[13] = 0.25f * s42 * x * (5.0f * z * z - 1.0f);
    sh[14] = 0.5f * s105 * z * (x * x - y * y);
    sh[15] = 0.25f * s70 * x * (x * x - 3.0f * y * y);
    #pragma unroll
    for (int i = 0; i < 16; ++i) sh_out[e * 16 + i] = sh[i];

    const float u = r * 0.2f;  // r / R_MAX
    float env = 0.0f;
    if (u < 1.0f) {
      const float u5 = u * u * u * u * u;
      env = 1.0f - 21.0f * u5 + 35.0f * u5 * u - 15.0f * u5 * u * u;
    }
    const float pre = 0.6324555320336759f / (r + 1e-8f);  // sqrt(2/5)/(r+eps)
    #pragma unroll
    for (int nn = 1; nn <= 8; ++nn)
      s_bess[eg][nn - 1] = pre * sinf((float)nn * 0.6283185307179586f * r) * env;

    float ed[16];
    diff_mlp(escal[e], W0, b0, W1, b1, W2, b2, ed);
    #pragma unroll
    for (int i = 0; i < 16; ++i) s_ediff[eg][i] = ed[i];
  }
  __syncthreads();

  float acc = 0.0f;
  #pragma unroll
  for (int i = 0; i < 16; ++i) {
    const float di = s_ediff[eg][i];
    #pragma unroll
    for (int j = 0; j < 8; ++j)
      acc += di * s_bess[eg][j] * emw[(i * 8 + j) * 64 + g];
  }
  s_ef[eg][g] = acc * 0.08838834764831845f;  // 1/sqrt(128)
  __syncthreads();

  float acc2 = 0.0f;
  for (int t2 = 0; t2 < 64; ++t2) acc2 += s_ef[eg][t2] * ehw[t2 * 64 + g];
  ef_out[e * 64 + g] = (_Float16)acc2;
}

// ---------------------------------------------------------------------------
// Message + scatter: msg[e,k,c] = sh[e,k]*w1*hs0 + w2*hs; atomic segment-sum.
// ---------------------------------------------------------------------------
__global__ __launch_bounds__(256)
void msg_scatter(const _Float16* __restrict__ w, const _Float16* __restrict__ hup,
                 const float* __restrict__ sh, const int* __restrict__ src,
                 const int* __restrict__ dst, float* __restrict__ agg, int E)
{
  const size_t t = (size_t)blockIdx.x * blockDim.x + threadIdx.x;
  const size_t e = t >> 6;
  const int c = (int)(t & 63);
  if (e >= (size_t)E) return;
  const int si = src[e], di = dst[e];
  const size_t hb = (size_t)si * 1024 + c;
  __builtin_prefetch(&hup[hb], 0, 0);  // global_prefetch_b8 on the gather stream
  const float hs0 = (float)hup[hb];
  const size_t wb = e * 512 + c;
  const size_t ab = (size_t)di * 1024 + c;
  #pragma unroll
  for (int k = 0; k < 16; ++k) {
    const int l = lof(k);
    const float w1 = (float)w[wb + l * 64];
    const float w2 = (float)w[wb + 256 + l * 64];
    const float hsk = (float)hup[hb + k * 64];
    const float msg = sh[e * 16 + k] * w1 * hs0 + w2 * hsk;
    unsafeAtomicAdd(&agg[ab + k * 64], msg * 0.0625f);  // /AVG_NEI
  }
}

// ---------------------------------------------------------------------------
// norm_tanh per l-slice + bfeat = m*(1+m0), gated by prod_z. Writes f16.
// ---------------------------------------------------------------------------
DEV void norm_slice(float* v, int a, int b)
{
  float s = 0.0f;
  for (int k = a; k < b; ++k) s += v[k] * v[k];
  const float n = sqrtf(s);
  const float f = tanhf(n) / (n + 1e-8f);
  for (int k = a; k < b; ++k) v[k] *= f;
}

__global__ void norm_prod(const float* __restrict__ m, const float* __restrict__ aug,
                          const float* __restrict__ pzW, _Float16* __restrict__ outh, int N)
{
  const size_t t = (size_t)blockIdx.x * blockDim.x + threadIdx.x;
  const int n = (int)(t >> 6);
  const int c = (int)(t & 63);
  if (n >= N) return;
  const size_t base = (size_t)n * 1024 + c;
  float v[16];
  #pragma unroll
  for (int k = 0; k < 16; ++k) v[k] = m[base + k * 64];
  norm_slice(v, 0, 1);
  norm_slice(v, 1, 4);
  norm_slice(v, 4, 9);
  norm_slice(v, 9, 16);
  const float m0 = v[0];
  float wz[4];
  #pragma unroll
  for (int l = 0; l < 4; ++l) {
    float a = 0.0f;
    #pragma unroll
    for (int z = 0; z < 5; ++z)
      a += aug[(size_t)n * 5 + z] * pzW[(l * 5 + z) * 64 + c];
    wz[l] = a;
  }
  #pragma unroll
  for (int k = 0; k < 16; ++k)
    outh[base + k * 64] = (_Float16)(v[k] * (1.0f + m0) * wz[lof(k)]);
}

// out[n] = [feats[n,1:4,:]@vec_W (3), feats[n,0,:]@cls_W (5)]
__global__ void final_out(const float* __restrict__ feats, const float* __restrict__ vecW,
                          const float* __restrict__ clsW, float* __restrict__ out, int N)
{
  const size_t t = (size_t)blockIdx.x * blockDim.x + threadIdx.x;
  const int n = (int)(t >> 3);
  const int o = (int)(t & 7);
  if (n >= N) return;
  float acc = 0.0f;
  if (o < 3) {
    const float* f = feats + (size_t)n * 1024 + (1 + o) * 64;
    for (int c = 0; c < 64; ++c) acc += f[c] * vecW[c];
  } else {
    const int z = o - 3;
    const float* f = feats + (size_t)n * 1024;
    for (int c = 0; c < 64; ++c) acc += f[c] * clsW[c * 5 + z];
  }
  out[(size_t)n * 8 + o] = acc;
}

// ---------------------------------------------------------------------------
extern "C" void kernel_launch(void* const* d_in, const int* in_sizes, int n_in,
                              void* d_out, int out_size, void* d_ws, size_t ws_size,
                              hipStream_t stream)
{
  (void)n_in; (void)out_size; (void)ws_size;
  const float* positions = (const float*)d_in[0];
  const float* node_attrs = (const float*)d_in[1];
  const float* nscal = (const float*)d_in[2];
  const float* escal = (const float*)d_in[3];
  // d_in[4] = forces (unused by the reference forward)
  const float* shifts = (const float*)d_in[5];
  const int* eidx = (const int*)d_in[6];
  const int N = in_sizes[0] / 3;
  const int E = in_sizes[5] / 3;
  const int* src = eidx;
  const int* dst = eidx + E;

  const float* dW0 = (const float*)d_in[7];
  const float* db0 = (const float*)d_in[8];
  const float* dW1 = (const float*)d_in[9];
  const float* db1 = (const float*)d_in[10];
  const float* dW2 = (const float*)d_in[11];
  const float* db2 = (const float*)d_in[12];
  const float* amw = (const float*)d_in[13];
  const float* embW = (const float*)d_in[14];
  const float* emw = (const float*)d_in[15];
  const float* ehw = (const float*)d_in[16];
  const float* vecW = (const float*)d_in[17];
  const float* clsW = (const float*)d_in[18];

  char* base = (char*)d_ws;
  size_t off = 0;
  auto alloc = [&](size_t bytes) -> void* {
    off = (off + 63) & ~(size_t)63;
    void* p = base + off;
    off += bytes;
    return p;
  };

  float*    aug    = (float*)alloc((size_t)N * 5 * 4);
  float*    feats  = (float*)alloc((size_t)N * 1024 * 4);
  _Float16* featsh = (_Float16*)alloc((size_t)N * 1024 * 2);
  float*    shb    = (float*)alloc((size_t)E * 16 * 4);
  _Float16* efh    = (_Float16*)alloc((size_t)E * 64 * 2);
  _Float16* gA     = (_Float16*)alloc((size_t)E * 64 * 2);
  _Float16* gB     = (_Float16*)alloc((size_t)E * 64 * 2);
  _Float16* wbuf   = (_Float16*)alloc((size_t)E * 512 * 2);  // also hosts A' (16*N*320 <= E*512)
  float*    scb    = (float*)alloc((size_t)N * 1024 * 4);
  float*    agg    = (float*)alloc((size_t)N * 1024 * 4);
  _Float16* wpk    = (_Float16*)alloc((size_t)2 * 176128 * 2);  // packed weights, both layers

  _Float16* aprime = wbuf;      // alias: A' built+consumed before edge MLP writes wbuf
  _Float16* huph   = gA;        // alias: free after edge MLP chain
  _Float16* tmph   = gB;        // alias: aggh then bfeat-f16
  float*    mb     = agg;       // alias: m overwrites agg f32 after aggh conversion

  auto rp = [&](const float* W, _Float16* Wp, int K, int NC, int nmat) {
    size_t per = (size_t)(NC / 16) * (K / 32) * 512;
    size_t total = per * nmat;
    repack_w<<<(int)((total + 255) / 256), 256, 0, stream>>>(W, Wp, K, NC, nmat);
  };

  // Repack all layer weights into WMMA fragment order (f16).
  for (int layer = 0; layer < 2; ++layer) {
    const int pb = 19 + 9 * layer;
    _Float16* L = wpk + (size_t)layer * 176128;
    rp((const float*)d_in[pb + 2], L + 0,      64, 64, 1);    // r0
    rp((const float*)d_in[pb + 3], L + 4096,   64, 64, 1);    // r1
    rp((const float*)d_in[pb + 4], L + 8192,   64, 64, 1);    // r2
    rp((const float*)d_in[pb + 5], L + 12288,  64, 512, 1);   // r3
    rp((const float*)d_in[pb + 1], L + 45056,  64, 64, 4);    // up_W
    rp((const float*)d_in[pb + 6], L + 61440,  64, 64, 4);    // out_W
    rp((const float*)d_in[pb + 8], L + 77824,  64, 64, 4);    // prod_lin_W
    rp((const float*)d_in[pb + 0], L + 94208, 320, 64, 4);    // sc_W
  }

  node_prep<<<(N + 255) / 256, 256, 0, stream>>>(nscal, node_attrs, dW0, db0, dW1, db1,
                                                 dW2, db2, amw, aug, N);
  feats_init<<<(N * 64) / 256, 256, 0, stream>>>(aug, embW, feats, featsh, N);
  edge_prep<<<(E + 3) / 4, 256, 0, stream>>>(positions, shifts, escal, src, dst,
                                             dW0, db0, dW1, db1, dW2, db2,
                                             emw, ehw, shb, efh, E);

  for (int layer = 0; layer < 2; ++layer) {
    const int pb = 19 + 9 * layer;
    const float* pzW = (const float*)d_in[pb + 7];
    _Float16* L = wpk + (size_t)layer * 176128;
    _Float16 *r0p = L, *r1p = L + 4096, *r2p = L + 8192, *r3p = L + 12288;
    _Float16 *upp = L + 45056, *outp = L + 61440, *plinp = L + 77824, *scp = L + 94208;

    dim3 gN(N / 16, 16);
    // sc from layer-start feats (before wbuf/A' aliasing conflict)
    build_aprime<<<(16 * N * 64) / 256, 256, 0, stream>>>(featsh, aug, aprime, N);
    wmma_sc_h<<<gN, 32, 0, stream>>>(aprime, scp, scb, N);

    // edge weight MLP chain (all f16, packed-B WMMA)
    dim3 gE(E / 16, 1);
    wmma_gemm_h<<<gE, 32, 0, stream>>>(efh, r0p, gA, nullptr, 64, 64, 1);
    wmma_gemm_h<<<gE, 32, 0, stream>>>(gA, r1p, gB, nullptr, 64, 64, 1);
    wmma_gemm_h<<<gE, 32, 0, stream>>>(gB, r2p, gA, nullptr, 64, 64, 1);
    dim3 gE3(E / 16, 8);
    wmma_gemm_h<<<gE3, 32, 0, stream>>>(gA, r3p, wbuf, nullptr, 64, 512, 0);

    wmma_nodek_h<<<gN, 32, 0, stream>>>(featsh, upp, nullptr, nullptr, huph);
    zero_kernel<<<2048, 256, 0, stream>>>(agg, (size_t)N * 1024);
    msg_scatter<<<(E * 64) / 256, 256, 0, stream>>>(wbuf, huph, shb, src, dst, agg, E);
    f32_to_f16<<<2048, 256, 0, stream>>>(agg, tmph, (size_t)N * 1024);
    wmma_nodek_h<<<gN, 32, 0, stream>>>(tmph, outp, nullptr, mb, nullptr);
    norm_prod<<<(N * 64) / 256, 256, 0, stream>>>(mb, aug, pzW, tmph, N);
    wmma_nodek_h<<<gN, 32, 0, stream>>>(tmph, plinp, scb, feats, featsh);
  }

  final_out<<<(N * 8) / 256, 256, 0, stream>>>(feats, vecW, clsW, (float*)d_out, N);
}
